// GraphConvNet_69157563400849
// MI455X (gfx1250) — compile-verified, matrix operand order checked by reading, not run on hardware
//
#include <hip/hip_runtime.h>
#include <stdint.h>

// ---------------------------------------------------------------- types
typedef __attribute__((ext_vector_type(16))) __bf16 v16bf;
typedef __attribute__((ext_vector_type(8)))  float  v8f;

#define LAT 128

// ---------------------------------------------------------------- helpers
static __device__ __forceinline__ __bf16 f2bf(float f) {
    unsigned u = __builtin_bit_cast(unsigned, f);
    unsigned r = u + 0x7fffu + ((u >> 16) & 1u);        // round-to-nearest-even
    unsigned short h = (unsigned short)(r >> 16);
    return __builtin_bit_cast(__bf16, h);
}

// branch-free tanh-approx gelu (matches jax.nn.gelu approximate=True)
static __device__ __forceinline__ float gelu(float x) {
    float u = 0.7978845608f * (x + 0.044715f * x * x * x);
#if __has_builtin(__builtin_amdgcn_tanhf)
    float th = __builtin_amdgcn_tanhf(u);               // v_tanh_f32 (gfx1250)
#else
    // tanh(u) = sign(u) * (1 - t) / (1 + t), t = exp(-2|u|) = exp2(-2*log2e*|u|)
    float au = __builtin_fabsf(u);
    float t  = __builtin_amdgcn_exp2f(-2.8853900818f * au);   // v_exp_f32
    float th = (1.0f - t) * __builtin_amdgcn_rcpf(1.0f + t);  // v_rcp_f32
    th = __builtin_copysignf(th, u);
#endif
    return 0.5f * x * (1.0f + th);
}

static __device__ __forceinline__ v8f wmma_bf16(v16bf a, v16bf b, v8f c) {
    // D = A(16xK) * B(Kx16) + C, emits v_wmma_f32_16x16x32_bf16
    return __builtin_amdgcn_wmma_f32_16x16x32_bf16(
        /*neg_a=*/false, a, /*neg_b=*/false, b,
        /*c_mod=*/(short)0, c, /*reuse_a=*/false, /*reuse_b=*/false);
}

// A-fragment (16x32 bf16 tile) stored contiguously per lane: chunk kc at kc*512.
static __device__ __forceinline__ v16bf afrag(const __bf16* buf, int lane, int kc) {
    return ((const v16bf*)(buf + kc * 512))[lane];
}
// B-fragment from packed global weights: block (kc,nc) -> 512 bf16.
static __device__ __forceinline__ v16bf bfrag(const __bf16* Wp, int lane, int kc, int nc) {
    return ((const v16bf*)Wp)[(size_t)(kc * 8 + nc) * 32 + lane];
}
// init C fragment with per-column bias
static __device__ __forceinline__ v8f cinit(const float* b, int lane, int nc) {
    float v = b[nc * 16 + (lane & 15)];
    v8f c;
#pragma unroll
    for (int i = 0; i < 8; ++i) c[i] = v;
    return c;
}
// scatter D fragment (16x16 f32, col block nc of a 16x128 Y) into LDS in
// A-fragment order for the *next* layer (K-dim = column index), as bf16.
static __device__ __forceinline__ void cfrag_to_lds(__bf16* buf, v8f c, int lane, int nc) {
    int ln = lane & 15, hf = lane >> 4;
    int n  = nc * 16 + ln;
    int kc = n >> 5, k5 = n & 31;
    int halfk = (k5 >> 3) & 1;
    int vv    = ((k5 & 7) >> 1) + ((k5 >> 4) << 2);
    int pair  = n & 1;
    int base  = kc * 512 + 16 * halfk * 16 + vv * 2 + pair;
#pragma unroll
    for (int r = 0; r < 8; ++r) {
        int m = hf * 8 + r;
        buf[base + m * 16] = f2bf(c[r]);
    }
}

// hidden layer: Y = gelu(X @ W + b), X in ldsIn (NKC k-chunks), Y -> ldsOut
template <int NKC>
static __device__ __forceinline__ void layer_hidden(const __bf16* ldsIn,
                                                    const __bf16* Wp, const float* bias,
                                                    __bf16* ldsOut, int lane) {
#pragma unroll
    for (int nc = 0; nc < 8; ++nc) {
        v8f c = cinit(bias, lane, nc);
#pragma unroll
        for (int kc = 0; kc < NKC; ++kc)
            c = wmma_bf16(afrag(ldsIn, lane, kc), bfrag(Wp, lane, kc, nc), c);
#pragma unroll
        for (int i = 0; i < 8; ++i) c[i] = gelu(c[i]);
        cfrag_to_lds(ldsOut, c, lane, nc);
    }
}

// ---------------------------------------------------------------- weight pack
// fp32 W[K][128] -> bf16 in WMMA B-fragment order (zero-pad K to Kpad)
__global__ __launch_bounds__(256) void pack_w_kernel(const float* __restrict__ W,
                                                     int K, int Kpad, __bf16* __restrict__ dst) {
    int idx = blockIdx.x * 256 + threadIdx.x;
    if (idx >= Kpad * LAT) return;
    int k = idx >> 7, n = idx & 127;
    float v = (k < K) ? W[(size_t)k * LAT + n] : 0.0f;
    int kc = k >> 5, k5 = k & 31;
    int halfk = (k5 >> 3) & 1;
    int vv    = ((k5 & 7) >> 1) + ((k5 >> 4) << 2);
    int pair  = k5 & 1;
    dst[(size_t)(kc * 8 + (n >> 4)) * 512 + ((n & 15) + 16 * halfk) * 16 + vv * 2 + pair] = f2bf(v);
}

// fold the (constant, single-graph) globals contribution into the bias
__global__ __launch_bounds__(128) void bias_fold_kernel(const float* __restrict__ b,
                                                        const float* __restrict__ Wg,
                                                        const float* __restrict__ g,
                                                        int grows, float* __restrict__ out) {
    int j = threadIdx.x;
    float v = b[j];
    for (int t = 0; t < grows; ++t) v += g[t] * Wg[(size_t)t * LAT + j];
    out[j] = v;
}

// ---------------------------------------------------------------- embed MLP
// X[M,KIN] -> L0(KIN->128, valu, gelu) -> L1(wmma, gelu) -> L2(wmma) -> Ylat
__global__ __launch_bounds__(256) void embed_mlp_kernel(
    const float* __restrict__ X, int KIN,
    const float* __restrict__ W0, const float* __restrict__ b0,
    const __bf16* __restrict__ W1p, const float* __restrict__ b1,
    const __bf16* __restrict__ W2p, const float* __restrict__ b2,
    float* __restrict__ Ylat, int ntiles) {
    __shared__ __attribute__((aligned(32))) __bf16 sA[8][4 * 512];
    __shared__ __attribute__((aligned(32))) __bf16 sB[8][4 * 512];
    int wave = threadIdx.x >> 5, lane = threadIdx.x & 31;
    int tile = blockIdx.x * 8 + wave;
    bool active = tile < ntiles;
    int tc = active ? tile : (ntiles - 1);
    int ln = lane & 15, hf = lane >> 4;
    __bf16* myA = sA[wave];
    __bf16* myB = sB[wave];

    // ---- L0 via VALU (KIN <= 6)
    float xr[8][6];
#pragma unroll
    for (int r = 0; r < 8; ++r) {
        size_t row = (size_t)tc * 16 + hf * 8 + r;
        for (int k = 0; k < KIN; ++k) xr[r][k] = X[row * KIN + k];
    }
#pragma unroll
    for (int nc = 0; nc < 8; ++nc) {
        int n = nc * 16 + ln;
        float wc[6];
        for (int k = 0; k < KIN; ++k) wc[k] = W0[(size_t)k * LAT + n];
        float bb = b0[n];
        v8f c;
#pragma unroll
        for (int r = 0; r < 8; ++r) {
            float a = bb;
            for (int k = 0; k < KIN; ++k) a += xr[r][k] * wc[k];
            c[r] = gelu(a);
        }
        cfrag_to_lds(myA, c, lane, nc);
    }
    __syncthreads();
    // ---- L1 (wmma)
    layer_hidden<4>(myA, W1p, b1, myB, lane);
    __syncthreads();
    // ---- L2 (wmma, linear) + store
#pragma unroll
    for (int nc = 0; nc < 8; ++nc) {
        v8f c = cinit(b2, lane, nc);
#pragma unroll
        for (int kc = 0; kc < 4; ++kc)
            c = wmma_bf16(afrag(myB, lane, kc), bfrag(W2p, lane, kc, nc), c);
        if (active) {
            int n = nc * 16 + ln;
#pragma unroll
            for (int r = 0; r < 8; ++r) {
                size_t row = (size_t)tile * 16 + hf * 8 + r;
                Ylat[row * LAT + n] = c[r];
            }
        }
    }
}

// ---------------------------------------------------------------- edge step
// in = [edge_lat | nodes[snd] | nodes[rcv]] (K=384, globals folded into b0)
// 3-layer MLP; then edges += out (residual) and atomic scatter-add to received.
__global__ __launch_bounds__(128) void edge_step_kernel(
    float* __restrict__ edges_lat, const float* __restrict__ nodes_lat,
    const int* __restrict__ senders, const int* __restrict__ receivers,
    float* __restrict__ received,
    const __bf16* __restrict__ W0p, const float* __restrict__ b0,
    const __bf16* __restrict__ W1p, const float* __restrict__ b1,
    const __bf16* __restrict__ W2p, const float* __restrict__ b2,
    int ntiles) {
    __shared__ __attribute__((aligned(32))) __bf16 sA[4][12 * 512];
    __shared__ __attribute__((aligned(32))) __bf16 sB[4][4 * 512];
    int wave = threadIdx.x >> 5, lane = threadIdx.x & 31;
    int tile = blockIdx.x * 4 + wave;
    bool active = tile < ntiles;
    int tc = active ? tile : (ntiles - 1);
    __bf16* myA = sA[wave];
    __bf16* myB = sB[wave];

    // ---- build 16x384 bf16 input tile in A-fragment order (float4 gathers)
    for (int i = lane; i < 16 * 96; i += 32) {
        int m = i / 96, q = i - m * 96;
        int k4 = q * 4;
        size_t e = (size_t)tc * 16 + m;
        const float* src;
        int kk;
        if (k4 < 128)      { src = edges_lat + e * LAT;                        kk = k4; }
        else if (k4 < 256) { src = nodes_lat + (size_t)senders[e] * LAT;       kk = k4 - 128; }
        else               { src = nodes_lat + (size_t)receivers[e] * LAT;     kk = k4 - 256; }
        float4 v = *(const float4*)(src + kk);
        int kc = k4 >> 5, k5 = k4 & 31;
        int halfk = (k5 >> 3) & 1;
        int vv    = ((k5 & 7) >> 1) + ((k5 >> 4) << 2);
        int off = kc * 512 + (m + 16 * halfk) * 16 + vv * 2;
        myA[off + 0] = f2bf(v.x); myA[off + 1] = f2bf(v.y);
        myA[off + 2] = f2bf(v.z); myA[off + 3] = f2bf(v.w);
    }
    __syncthreads();
    layer_hidden<12>(myA, W0p, b0, myB, lane);   // 388->128 (globals folded)
    __syncthreads();
    layer_hidden<4>(myB, W1p, b1, myA, lane);    // 128->128
    __syncthreads();
    // ---- final linear + residual + scatter
    int ln = lane & 15, hf = lane >> 4;
#pragma unroll
    for (int nc = 0; nc < 8; ++nc) {
        v8f c = cinit(b2, lane, nc);
#pragma unroll
        for (int kc = 0; kc < 4; ++kc)
            c = wmma_bf16(afrag(myA, lane, kc), bfrag(W2p, lane, kc, nc), c);
        if (active) {
            int n = nc * 16 + ln;
#pragma unroll
            for (int r = 0; r < 8; ++r) {
                size_t e = (size_t)tile * 16 + hf * 8 + r;
                float v = c[r];
                float* ep = edges_lat + e * LAT + n;
                *ep = *ep + v;                               // edge residual
                int rid = receivers[e];
                __hip_atomic_fetch_add(received + (size_t)rid * LAT + n, v,
                                       __ATOMIC_RELAXED, __HIP_MEMORY_SCOPE_AGENT);
            }
        }
    }
}

// ---------------------------------------------------------------- node step
// in = [nodes | received] (K=256, globals folded into b0); nodes += MLP(in)
__global__ __launch_bounds__(128) void node_step_kernel(
    float* __restrict__ nodes_lat, const float* __restrict__ received,
    const __bf16* __restrict__ W0p, const float* __restrict__ b0,
    const __bf16* __restrict__ W1p, const float* __restrict__ b1,
    const __bf16* __restrict__ W2p, const float* __restrict__ b2,
    int ntiles) {
    __shared__ __attribute__((aligned(32))) __bf16 sA[4][8 * 512];
    __shared__ __attribute__((aligned(32))) __bf16 sB[4][4 * 512];
    int wave = threadIdx.x >> 5, lane = threadIdx.x & 31;
    int tile = blockIdx.x * 4 + wave;
    bool active = tile < ntiles;
    int tc = active ? tile : (ntiles - 1);
    __bf16* myA = sA[wave];
    __bf16* myB = sB[wave];

    for (int i = lane; i < 16 * 64; i += 32) {
        int m = i >> 6, q = i & 63;
        int k4 = q * 4;
        size_t row = (size_t)tc * 16 + m;
        const float* src = (k4 < 128) ? (nodes_lat + row * LAT) : (received + row * LAT);
        int kk = (k4 < 128) ? k4 : (k4 - 128);
        float4 v = *(const float4*)(src + kk);
        int kc = k4 >> 5, k5 = k4 & 31;
        int halfk = (k5 >> 3) & 1;
        int vv    = ((k5 & 7) >> 1) + ((k5 >> 4) << 2);
        int off = kc * 512 + (m + 16 * halfk) * 16 + vv * 2;
        myA[off + 0] = f2bf(v.x); myA[off + 1] = f2bf(v.y);
        myA[off + 2] = f2bf(v.z); myA[off + 3] = f2bf(v.w);
    }
    __syncthreads();
    layer_hidden<8>(myA, W0p, b0, myB, lane);
    __syncthreads();
    layer_hidden<4>(myB, W1p, b1, myA, lane);
    __syncthreads();
    int ln = lane & 15, hf = lane >> 4;
#pragma unroll
    for (int nc = 0; nc < 8; ++nc) {
        v8f c = cinit(b2, lane, nc);
#pragma unroll
        for (int kc = 0; kc < 4; ++kc)
            c = wmma_bf16(afrag(myA, lane, kc), bfrag(W2p, lane, kc, nc), c);
        if (active) {
            int n = nc * 16 + ln;
#pragma unroll
            for (int r = 0; r < 8; ++r) {
                size_t row = (size_t)tile * 16 + hf * 8 + r;
                float* np = nodes_lat + row * LAT + n;
                *np = *np + c[r];                            // node residual
            }
        }
    }
}

// ---------------------------------------------------------------- launch
extern "C" void kernel_launch(void* const* d_in, const int* in_sizes, int n_in,
                              void* d_out, int out_size, void* d_ws, size_t ws_size,
                              hipStream_t stream) {
    const float* nodes_feat   = (const float*)d_in[0];
    const float* edges_feat   = (const float*)d_in[1];
    const float* globals_feat = (const float*)d_in[2];
    const int*   senders      = (const int*)d_in[3];
    const int*   receivers    = (const int*)d_in[4];
    // param flattening: node_embed 5..10, edge_embed 11..16,
    // edge_step 17..34 (3 steps x 3 layers x (W,b)), node_step 35..52
    const int N = in_sizes[0] / 6;
    const int E = in_sizes[1] / 3;

    size_t off = 0;
    auto alloc = [&](size_t bytes) -> void* {
        void* p = (char*)d_ws + off;
        off += (bytes + 255) & ~(size_t)255;
        return p;
    };
    float* edges_lat = (float*)alloc((size_t)E * LAT * 4);
    float* received  = (float*)alloc((size_t)N * LAT * 4);
    float* nodes_lat = (float*)d_out;

    auto packW = [&](int widx, int K) -> __bf16* {
        int Kpad = ((K + 31) / 32) * 32;
        __bf16* dst = (__bf16*)alloc((size_t)Kpad * LAT * 2);
        int total = Kpad * LAT;
        pack_w_kernel<<<(total + 255) / 256, 256, 0, stream>>>(
            (const float*)d_in[widx], K, Kpad, dst);
        return dst;
    };
    auto foldB = [&](int widx, int bidx, int K) -> float* {
        float* dst = (float*)alloc(LAT * 4);
        bias_fold_kernel<<<1, 128, 0, stream>>>(
            (const float*)d_in[bidx], (const float*)d_in[widx] + (size_t)K * LAT,
            globals_feat, 4, dst);
        return dst;
    };

    // embed nets (L0 handled in-kernel with raw fp32 weights)
    __bf16* neW1 = packW(7, 128);
    __bf16* neW2 = packW(9, 128);
    __bf16* eeW1 = packW(13, 128);
    __bf16* eeW2 = packW(15, 128);
    // step nets
    __bf16 *esW0[3], *esW1[3], *esW2[3], *nsW0[3], *nsW1[3], *nsW2[3];
    float  *esB0[3], *nsB0[3];
    for (int s = 0; s < 3; ++s) {
        int eb = 17 + 6 * s, nb = 35 + 6 * s;
        esW0[s] = packW(eb, 384);          // first 384 rows; globals rows folded:
        esB0[s] = foldB(eb, eb + 1, 384);
        esW1[s] = packW(eb + 2, 128);
        esW2[s] = packW(eb + 4, 128);
        nsW0[s] = packW(nb, 256);
        nsB0[s] = foldB(nb, nb + 1, 256);
        nsW1[s] = packW(nb + 2, 128);
        nsW2[s] = packW(nb + 4, 128);
    }

    const int nodeTiles = N / 16;   // 3125
    const int edgeTiles = E / 16;   // 25000

    // embedders
    embed_mlp_kernel<<<(nodeTiles + 7) / 8, 256, 0, stream>>>(
        nodes_feat, 6, (const float*)d_in[5], (const float*)d_in[6],
        neW1, (const float*)d_in[8], neW2, (const float*)d_in[10],
        nodes_lat, nodeTiles);
    embed_mlp_kernel<<<(edgeTiles + 7) / 8, 256, 0, stream>>>(
        edges_feat, 3, (const float*)d_in[11], (const float*)d_in[12],
        eeW1, (const float*)d_in[14], eeW2, (const float*)d_in[16],
        edges_lat, edgeTiles);

    // message-passing steps
    for (int s = 0; s < 3; ++s) {
        int eb = 17 + 6 * s, nb = 35 + 6 * s;
        hipMemsetAsync(received, 0, (size_t)N * LAT * 4, stream);
        edge_step_kernel<<<(edgeTiles + 3) / 4, 128, 0, stream>>>(
            edges_lat, nodes_lat, senders, receivers, received,
            esW0[s], esB0[s], esW1[s], (const float*)d_in[eb + 3],
            esW2[s], (const float*)d_in[eb + 5], edgeTiles);
        node_step_kernel<<<(nodeTiles + 3) / 4, 128, 0, stream>>>(
            nodes_lat, received,
            nsW0[s], nsB0[s], nsW1[s], (const float*)d_in[nb + 3],
            nsW2[s], (const float*)d_in[nb + 5], nodeTiles);
    }
    (void)n_in; (void)out_size; (void)ws_size;
}
